// MainNetwork_41128606826605
// MI455X (gfx1250) — matrix operand order repaired and checked by reading, not compiled, not run on hardware
//
#include <hip/hip_runtime.h>
#include <hip/hip_bf16.h>

typedef __attribute__((ext_vector_type(16))) _Float16 v16h;
typedef __attribute__((ext_vector_type(8)))  float    v8f;

#define LENA   8
#define BATCH  128
#define ARR    1024
#define HID    64
#define XCHUNK 256
#define HSTRIDE 80   // padded f16 row stride: 160B rows -> 16B-aligned fragment reads

__device__ __forceinline__ float fast_sigmoid(float v) {
  return __builtin_amdgcn_rcpf(1.0f + __expf(-v));
}
__device__ __forceinline__ float fast_tanh(float v) {
  // tanh(x) = 2*sigmoid(2x) - 1
  return __builtin_amdgcn_rcpf(1.0f + __expf(-2.0f * v)) * 2.0f - 1.0f;
}

// One workgroup per (batch-tile of 16, LSTM l). 4 wave32 waves; wave w owns
// hidden slice [16w, 16w+16) across all four gates, so the cell update is
// wave-local. h is exchanged via double-buffered LDS once per step.
__global__ __launch_bounds__(128) void lstm_ensemble_kernel(
    const float* __restrict__ x,   // (B, ARR)
    const float* __restrict__ hn,  // (L, B, H)
    const float* __restrict__ Wih, // (L, 4H, 1)
    const float* __restrict__ Whh, // (L, 4H, H)
    const float* __restrict__ bih, // (L, 4H)
    const float* __restrict__ bhh, // (L, 4H)
    float* __restrict__ ysws)      // (L, 8, ARR, 16)
{
  __shared__ float    xs[XCHUNK * 16];
  __shared__ _Float16 hbuf[2][16 * HSTRIDE];

  const int btile = blockIdx.x;
  const int l     = blockIdx.y;
  const int tid   = threadIdx.x;
  const int w     = tid >> 5;     // wave id 0..3
  const int lane  = tid & 31;
  const int half  = lane >> 4;    // lane-half (K / M split per ISA layouts)
  const int n     = lane & 15;

  // Stage initial h (f32 -> f16) into buffer 0.
  for (int idx = tid; idx < 16 * HID; idx += 128) {
    int b = idx >> 6, hid = idx & 63;
    hbuf[0][b * HSTRIDE + hid] =
        (_Float16)hn[((size_t)l * BATCH + btile * 16 + b) * HID + hid];
  }

  // Preload step-invariant B fragments (Whh^T tiles) and per-column scalars.
  // B 32x16 layout: lane n = column (gate col), element e -> K = half*16 + e.
  v16h Bf[4][2];
  float wih_s[4], bias_s[4];
#pragma unroll
  for (int G = 0; G < 4; ++G) {
    const int gcol = G * 64 + w * 16 + n;
    const float* wrow = Whh + ((size_t)l * 256 + gcol) * HID;
#pragma unroll
    for (int kc = 0; kc < 2; ++kc) {
      const float* src = wrow + kc * 32 + half * 16;
      v16h f;
#pragma unroll
      for (int e = 0; e < 16; ++e) f[e] = (_Float16)src[e];
      Bf[G][kc] = f;
    }
    wih_s[G]  = Wih[(size_t)l * 256 + gcol];
    bias_s[G] = bih[(size_t)l * 256 + gcol] + bhh[(size_t)l * 256 + gcol];
  }

  float c_state[8];
#pragma unroll
  for (int r = 0; r < 8; ++r) c_state[r] = 0.0f;

  int p = 0;
  for (int t = 0; t < ARR; ++t) {
    if ((t & (XCHUNK - 1)) == 0) {
      // Re-stage an x chunk (coalesced along t); prior reads are already
      // fenced by the end-of-step barrier.
      for (int idx = tid; idx < 16 * XCHUNK; idx += 128) {
        int tl = idx & (XCHUNK - 1);
        int b  = idx >> 8;
        xs[tl * 16 + b] = x[((size_t)(btile * 16 + b)) * ARR + t + tl];
      }
      __syncthreads();
    }

    // A fragments from shared h: lane row M = n; element e holds
    // K = kchunk*32 + half*16 + e (matches B).
    const _Float16* hb = &hbuf[p][0];
    v16h A0, A1;
    {
      const int base = n * HSTRIDE + half * 16;
#pragma unroll
      for (int e = 0; e < 16; ++e) {
        A0[e]     = hb[base + e];
        A1[e]     = hb[base + 32 + e];
      }
    }

    float xv[8];
    const int tl = t & (XCHUNK - 1);
#pragma unroll
    for (int r = 0; r < 8; ++r) xv[r] = xs[tl * 16 + r + 8 * half];

    // gates = x_t * Wih + bias  (rank-1 C init)  +  h @ Whh^T  (WMMA)
    v8f acc[4];
#pragma unroll
    for (int G = 0; G < 4; ++G) {
      v8f cini;
#pragma unroll
      for (int r = 0; r < 8; ++r) cini[r] = fmaf(xv[r], wih_s[G], bias_s[G]);
      v8f mid = __builtin_amdgcn_wmma_f32_16x16x32_f16(
          false, A0, false, Bf[G][0], (short)0, cini, false, false);
      acc[G] = __builtin_amdgcn_wmma_f32_16x16x32_f16(
          false, A1, false, Bf[G][1], (short)0, mid, false, false);
    }

    // Wave-local LSTM cell update. C/D layout: VGPR r -> batch row r + 8*half.
    float hnew[8];
#pragma unroll
    for (int r = 0; r < 8; ++r) {
      float ig = fast_sigmoid(acc[0][r]);
      float fg = fast_sigmoid(acc[1][r]);
      float gg = fast_tanh(acc[2][r]);
      float og = fast_sigmoid(acc[3][r]);
      float cc = fmaf(fg, c_state[r], ig * gg);
      c_state[r] = cc;
      hnew[r] = og * fast_tanh(cc);
    }

    // Publish h to the other buffer (hid column = 16w + n).
    _Float16* hw = &hbuf[p ^ 1][0];
#pragma unroll
    for (int r = 0; r < 8; ++r)
      hw[(r + 8 * half) * HSTRIDE + w * 16 + n] = (_Float16)hnew[r];

    // Last hidden unit (hid 63) lives in wave 3, column 15: stream to ws.
    if (w == 3 && n == 15) {
      float* dst = ysws + (((size_t)(l * 8 + btile) * ARR + t) << 4);
#pragma unroll
      for (int r = 0; r < 8; ++r) dst[r + 8 * half] = hnew[r];
    }

    __syncthreads();   // single barrier per step (double-buffered h)
    p ^= 1;
  }
}

// out[b][a] = bl[a] + sum_t (sum_l ys[l,b,t]) * Wl[a][t]
__global__ __launch_bounds__(128) void finalize_kernel(
    const float* __restrict__ ysws, const float* __restrict__ Wl,
    const float* __restrict__ bl, float* __restrict__ out)
{
  const int btile = blockIdx.x;
  const int b = threadIdx.x >> 3;
  const int a = threadIdx.x & 7;
  float acc = bl[a];
  for (int t = 0; t < ARR; ++t) {
    float r = 0.0f;
#pragma unroll
    for (int l = 0; l < 8; ++l)
      r += ysws[(((size_t)(l * 8 + btile) * ARR + t) << 4) + b];
    acc = fmaf(r, Wl[a * ARR + t], acc);
  }
  out[(btile * 16 + b) * 8 + a] = acc;
}

extern "C" void kernel_launch(void* const* d_in, const int* in_sizes, int n_in,
                              void* d_out, int out_size, void* d_ws, size_t ws_size,
                              hipStream_t stream) {
  const float* x   = (const float*)d_in[0];
  const float* hn  = (const float*)d_in[1];
  const float* Wih = (const float*)d_in[2];
  const float* Whh = (const float*)d_in[3];
  const float* bih = (const float*)d_in[4];
  const float* bhh = (const float*)d_in[5];
  const float* Wl  = (const float*)d_in[6];
  const float* bl  = (const float*)d_in[7];
  float* out  = (float*)d_out;
  float* ysws = (float*)d_ws;   // 8*8*1024*16 floats = 4 MiB

  lstm_ensemble_kernel<<<dim3(8, 8), 128, 0, stream>>>(x, hn, Wih, Whh, bih, bhh, ysws);
  finalize_kernel<<<dim3(8), 128, 0, stream>>>(ysws, Wl, bl, out);
}
